// DeepTime_59923383714504
// MI455X (gfx1250) — compile-verified, matrix operand order checked by reading, not run on hardware
//
#include <hip/hip_runtime.h>
#include <hip/hip_bf16.h>
#include <hip/hip_fp16.h>

// ---------------------------------------------------------------------------
// DeepTime (SIREN features + batched ridge regression) for MI455X / gfx1250.
// Dominant FLOPs (~240 GFLOP) are GEMM-shaped -> v_wmma_f32_16x16x32_f16
// (f16 operands, f32 accumulate). Tile staging uses the CDNA5 async
// memory->LDS path (global_load_async_to_lds_b128, ASYNCcnt) so the wave's
// issue slots stay on ds_load + wmma. Cholesky (<2% FLOPs) stays scalar fp32.
// ---------------------------------------------------------------------------

typedef __attribute__((ext_vector_type(16))) _Float16 v16h;
typedef __attribute__((ext_vector_type(8)))  float    v8f;

#define B_      64
#define NC_     1024
#define NT_     1024
#define H_      512
#define MROWS   (B_ * (NC_ + NT_))   // 131072 activation rows (ctx+tgt interleaved per batch)
#define COVLD   520                  // fp32 leading dim for 513x513 cov (8-float aligned)
#define XTYLD   528
#define OMEGA0  30.0f

// ---- CDNA5 async copy: memory -> LDS, 16B per lane, tracked by ASYNCcnt ----
__device__ __forceinline__ void async_copy_b128(void* lds_ptr, const void* gptr) {
  // LDS operand = low 32 bits of the flat shared address
  unsigned l = (unsigned)(uintptr_t)lds_ptr;
  asm volatile("global_load_async_to_lds_b128 %0, %1, off"
               :: "v"(l), "v"(gptr) : "memory");
}
__device__ __forceinline__ void wait_async() {
  asm volatile("s_wait_asynccnt 0x0" ::: "memory");
}

// ---------------------------------------------------------------- k_stats ---
__global__ __launch_bounds__(256)
void k_stats(const float* __restrict__ y, float* __restrict__ yc,
             float* __restrict__ meanstd) {
  const int b = blockIdx.x, t = threadIdx.x;
  __shared__ float ss[256], sq[256];
  __shared__ float smean, sstd;
  float s = 0.f, q = 0.f;
  for (int n = t; n < NC_; n += 256) {
    float v = y[b * NC_ + n];
    s += v; q += v * v;
  }
  ss[t] = s; sq[t] = q;
  __syncthreads();
  for (int o = 128; o > 0; o >>= 1) {
    if (t < o) { ss[t] += ss[t + o]; sq[t] += sq[t + o]; }
    __syncthreads();
  }
  if (t == 0) {
    float mean = ss[0] * (1.0f / NC_);
    float var  = (sq[0] - (float)NC_ * mean * mean) * (1.0f / (NC_ - 1)); // ddof=1
    float sd   = sqrtf(var);
    meanstd[2 * b] = mean; meanstd[2 * b + 1] = sd;
    smean = mean; sstd = sd;
  }
  __syncthreads();
  float mean = smean, inv = 1.0f / sstd;
  for (int n = t; n < NC_; n += 256)
    yc[b * NC_ + n] = (y[b * NC_ + n] - mean) * inv;
}

// ---------------------------------------------------------------- k_wconv ---
// Wt[l][n][k] = W_l[k][n] as f16 (transposed so B-fragments are contiguous).
__global__ __launch_bounds__(256)
void k_wconv(const float* __restrict__ W1, const float* __restrict__ W2,
             const float* __restrict__ Wr, _Float16* __restrict__ wt) {
  int idx = blockIdx.x * 256 + threadIdx.x;          // 3 * 512 * 512
  int l = idx >> 18;
  int rem = idx & ((1 << 18) - 1);
  int k = rem >> 9, n = rem & 511;
  const float* W = (l == 0) ? W1 : (l == 1) ? W2 : Wr;
  wt[(size_t)l * (H_ * H_) + (size_t)n * H_ + k] = (_Float16)W[k * H_ + n];
}

// --------------------------------------------------------------- k_layer0 ---
// Row mapping: row = b*2048 + i, i<1024 -> context point i, else target i-1024.
__global__ __launch_bounds__(256)
void k_layer0(const float* __restrict__ xc, const float* __restrict__ xt,
              const float* __restrict__ W0, const float* __restrict__ b0,
              _Float16* __restrict__ act) {
  long long idx = (long long)blockIdx.x * 256 + threadIdx.x;  // MROWS*512
  int row = (int)(idx >> 9), col = (int)(idx & 511);
  int b = row >> 11, i = row & 2047;
  float x = (i < NC_) ? xc[b * NC_ + i] : xt[b * NT_ + (i - NC_)];
  float h = sinf(OMEGA0 * (x * W0[col] + b0[col]));
  act[(size_t)row * H_ + col] = (_Float16)h;
}

// ------------------------------------------------------------- k_gemm_f16 ---
// C[M x 512] = act( A[M x 512] @ W + bias ), W given transposed f16 [512n x 512k].
// Workgroup tile 128(M) x 64(N), BK=32; 8 waves, each wave: 16 rows x 64 cols
// = 4 accumulators of v_wmma_f32_16x16x32_f16. Async-to-LDS staged operands.
__global__ __launch_bounds__(256)
void k_gemm_f16(const _Float16* __restrict__ A, const _Float16* __restrict__ Bt,
                const float* __restrict__ bias, _Float16* __restrict__ C,
                int applySin) {
  __shared__ alignas(32) _Float16 As[128 * 32];   // 8 KB
  __shared__ alignas(32) _Float16 Bs[64 * 32];    // 4 KB
  const int t = threadIdx.x, lane = t & 31, wid = t >> 5;
  const size_t rowBase = (size_t)blockIdx.y * 128;
  const int    colBase = blockIdx.x * 64;

  v8f acc[4] = {};
  const uint4* Ag = reinterpret_cast<const uint4*>(A);
  const uint4* Bg = reinterpret_cast<const uint4*>(Bt);

  for (int k0 = 0; k0 < H_; k0 += 32) {
    // stage A tile: 128 rows x 32 halves = 512 x b128 (2 per thread), async
#pragma unroll
    for (int p = 0; p < 2; ++p) {
      int q = t + p * 256;
      int row = q >> 2, seg = q & 3;
      async_copy_b128(&As[row * 32 + seg * 8],
                      &Ag[(rowBase + row) * (H_ / 8) + (k0 >> 3) + seg]);
    }
    // stage B tile: 64 rows(n) x 32 halves(k) = 256 x b128 (1 per thread)
    {
      int row = t >> 2, seg = t & 3;
      async_copy_b128(&Bs[row * 32 + seg * 8],
                      &Bg[(size_t)(colBase + row) * (H_ / 8) + (k0 >> 3) + seg]);
    }
    wait_async();
    __syncthreads();

    // A fragment: lane (m = lane&15) takes 16 consecutive halves; upper half-
    // wave takes the other 16 K values (contiguous k = 16h + s, same order as the B fragment).
    int mrow  = wid * 16 + (lane & 15);
    int khalf = (lane >> 4) * 16;
    v16h a = *reinterpret_cast<const v16h*>(&As[mrow * 32 + khalf]);
#pragma unroll
    for (int j = 0; j < 4; ++j) {
      int ncol = j * 16 + (lane & 15);
      v16h bf = *reinterpret_cast<const v16h*>(&Bs[ncol * 32 + khalf]);
      acc[j] = __builtin_amdgcn_wmma_f32_16x16x32_f16(
          false, a, false, bf, (short)0, acc[j], false, false);
    }
    __syncthreads();
  }

  // epilogue: C/D layout -> VGPR i holds (m = i + 8*(lane>=16), n = lane&15)
  const int mbase = wid * 16 + ((lane >> 4) << 3);
  const int nloc  = lane & 15;
#pragma unroll
  for (int j = 0; j < 4; ++j) {
    int col = colBase + j * 16 + nloc;
    float bj = bias[col];
#pragma unroll
    for (int i = 0; i < 8; ++i) {
      float v = acc[j][i] + bj;
      if (applySin) v = sinf(v);
      C[(rowBase + mbase + i) * (size_t)H_ + col] = (_Float16)v;
    }
  }
}

// ------------------------------------------------------------ k_transpose ---
// crT[b][r][n] = rep[b*2048 + n][r]  (context rows only), f16.
__global__ __launch_bounds__(256)
void k_transpose(const _Float16* __restrict__ rep, _Float16* __restrict__ crT) {
  long long idx = (long long)blockIdx.x * 256 + threadIdx.x;  // 64*512*1024
  int b = (int)(idx >> 19);
  int rem = (int)(idx & ((1 << 19) - 1));
  int r = rem >> 10, n = rem & 1023;
  crT[((size_t)b << 19) + (size_t)r * NC_ + n] =
      rep[((size_t)(b * 2048 + n)) * H_ + r];
}

// ----------------------------------------------------------------- k_syrk ---
// cov[b][0:512][0:512] = crT[b] (512 x 1024) @ crT[b]^T, via WMMA; fp32 out.
__global__ __launch_bounds__(256)
void k_syrk(const _Float16* __restrict__ crT, float* __restrict__ cov) {
  __shared__ alignas(32) _Float16 As[128 * 32];
  __shared__ alignas(32) _Float16 Bs[64 * 32];
  const int t = threadIdx.x, lane = t & 31, wid = t >> 5;
  const int b = blockIdx.z;
  const _Float16* Ab = crT + ((size_t)b << 19);   // 512 x 1024
  const size_t rowBase = (size_t)blockIdx.y * 128;  // M in [0,512)
  const int    colBase = blockIdx.x * 64;           // N in [0,512)

  v8f acc[4] = {};
  const uint4* Ag = reinterpret_cast<const uint4*>(Ab);

  for (int k0 = 0; k0 < NC_; k0 += 32) {
#pragma unroll
    for (int p = 0; p < 2; ++p) {
      int q = t + p * 256;
      int row = q >> 2, seg = q & 3;
      async_copy_b128(&As[row * 32 + seg * 8],
                      &Ag[(rowBase + row) * (NC_ / 8) + (k0 >> 3) + seg]);
    }
    {
      int row = t >> 2, seg = t & 3;
      async_copy_b128(&Bs[row * 32 + seg * 8],
                      &Ag[(size_t)(colBase + row) * (NC_ / 8) + (k0 >> 3) + seg]);
    }
    wait_async();
    __syncthreads();

    int mrow  = wid * 16 + (lane & 15);
    int khalf = (lane >> 4) * 16;
    v16h a = *reinterpret_cast<const v16h*>(&As[mrow * 32 + khalf]);
#pragma unroll
    for (int j = 0; j < 4; ++j) {
      int ncol = j * 16 + (lane & 15);
      v16h bf = *reinterpret_cast<const v16h*>(&Bs[ncol * 32 + khalf]);
      acc[j] = __builtin_amdgcn_wmma_f32_16x16x32_f16(
          false, a, false, bf, (short)0, acc[j], false, false);
    }
    __syncthreads();
  }

  float* covb = cov + (size_t)b * 513 * COVLD;
  const int mbase = wid * 16 + ((lane >> 4) << 3);
  const int nloc  = lane & 15;
#pragma unroll
  for (int j = 0; j < 4; ++j) {
    int col = colBase + j * 16 + nloc;
#pragma unroll
    for (int i = 0; i < 8; ++i)
      covb[(rowBase + mbase + i) * COVLD + col] = acc[j][i];
  }
}

// --------------------------------------------------------------- k_border ---
// Ones-feature row/col of cov, xty = cr^T yc, diagonal noise.
__global__ __launch_bounds__(256)
void k_border(const _Float16* __restrict__ rep, const float* __restrict__ yc,
              float* __restrict__ cov, float* __restrict__ xty,
              const float* __restrict__ lognv) {
  const int b = blockIdx.y;
  const int r = blockIdx.x * 256 + threadIdx.x;
  if (r >= 513) return;
  const float nv = expf(lognv[0]);
  const _Float16* repb = rep + ((size_t)b * 2048) * H_;   // context rows
  const float* ycb = yc + b * NC_;
  float s = 0.f, sy = 0.f;
  if (r < 512) {
    for (int n = 0; n < NC_; ++n) {
      float v = (float)repb[(size_t)n * H_ + r];
      s += v; sy += v * ycb[n];
    }
  } else {
    for (int n = 0; n < NC_; ++n) sy += ycb[n];
    s = (float)NC_;
  }
  float* covb = cov + (size_t)b * 513 * COVLD;
  if (r < 512) {
    covb[512 * COVLD + r] = s;
    covb[r * COVLD + 512] = s;
    covb[r * COVLD + r]  += nv;
  } else {
    covb[512 * COVLD + 512] = s + nv;
  }
  xty[b * XTYLD + r] = sy;
}

// ----------------------------------------------------------- k_chol_solve ---
// One workgroup per batch. Right-looking fp32 Cholesky with full-square
// trailing updates (keeps the matrix symmetric -> no triangle logic), then
// L z = v and L^T w = z, all in place. <2% of total FLOPs.
__global__ __launch_bounds__(256)
void k_chol_solve(float* __restrict__ cov, float* __restrict__ xty) {
  const int b = blockIdx.x, t = threadIdx.x;
  const int N = 513;
  float* Cm = cov + (size_t)b * 513 * COVLD;
  float* v  = xty + b * XTYLD;
  __shared__ float sval;

  for (int k = 0; k < N; ++k) {
    if (t == 0) {
      float d = sqrtf(Cm[k * COVLD + k]);
      Cm[k * COVLD + k] = d;
      sval = 1.0f / d;
    }
    __syncthreads();
    float inv = sval;
    for (int i = k + 1 + t; i < N; i += 256) Cm[i * COVLD + k] *= inv;
    __syncthreads();
    int w = N - k - 1;
    for (int idx = t; idx < w * w; idx += 256) {
      int i = k + 1 + idx / w;
      int j = k + 1 + idx % w;
      Cm[i * COVLD + j] = fmaf(-Cm[i * COVLD + k], Cm[j * COVLD + k],
                               Cm[i * COVLD + j]);
    }
    __syncthreads();
  }
  // forward: L z = v
  for (int k = 0; k < N; ++k) {
    if (t == 0) { v[k] /= Cm[k * COVLD + k]; sval = v[k]; }
    __syncthreads();
    float zk = sval;
    for (int i = k + 1 + t; i < N; i += 256)
      v[i] = fmaf(-Cm[i * COVLD + k], zk, v[i]);
    __syncthreads();
  }
  // backward: L^T w = z   (L^T[i][k] = L[k][i])
  for (int k = N - 1; k >= 0; --k) {
    if (t == 0) { v[k] /= Cm[k * COVLD + k]; sval = v[k]; }
    __syncthreads();
    float wk = sval;
    for (int i = t; i < k; i += 256)
      v[i] = fmaf(-Cm[k * COVLD + i], wk, v[i]);
    __syncthreads();
  }
}

// ----------------------------------------------------------------- k_pred ---
__global__ __launch_bounds__(256)
void k_pred(const _Float16* __restrict__ rep, const float* __restrict__ w,
            const float* __restrict__ meanstd, float* __restrict__ out) {
  const int b = blockIdx.y;
  const int m = blockIdx.x * 256 + threadIdx.x;   // [0,1024)
  const _Float16* tr = rep + ((size_t)(b * 2048 + NC_ + m)) * H_;
  const float* wb = w + b * XTYLD;
  float s = wb[512];
  for (int r = 0; r < H_; ++r) s = fmaf((float)tr[r], wb[r], s);
  out[b * NT_ + m] = s * meanstd[2 * b + 1] + meanstd[2 * b];
}

// ----------------------------------------------------------------- launch ---
extern "C" void kernel_launch(void* const* d_in, const int* in_sizes, int n_in,
                              void* d_out, int out_size, void* d_ws, size_t ws_size,
                              hipStream_t stream) {
  (void)in_sizes; (void)n_in; (void)out_size; (void)ws_size;
  const float* xc  = (const float*)d_in[0];
  const float* yct = (const float*)d_in[1];
  const float* xt  = (const float*)d_in[2];
  const float* W0  = (const float*)d_in[3];
  const float* b0  = (const float*)d_in[4];
  const float* W1  = (const float*)d_in[5];
  const float* b1  = (const float*)d_in[6];
  const float* W2  = (const float*)d_in[7];
  const float* b2  = (const float*)d_in[8];
  const float* Wr  = (const float*)d_in[9];
  const float* br  = (const float*)d_in[10];
  const float* lnv = (const float*)d_in[11];
  float* out = (float*)d_out;

  // bump allocator on the workspace (needs ~341 MB)
  char* ws = (char*)d_ws;
  size_t off = 0;
  auto bump = [&](size_t bytes) -> char* {
    char* p = ws + off;
    off += (bytes + 255) & ~(size_t)255;
    return p;
  };
  _Float16* act0 = (_Float16*)bump((size_t)MROWS * H_ * sizeof(_Float16)); // 128 MiB
  _Float16* act1 = (_Float16*)bump((size_t)MROWS * H_ * sizeof(_Float16)); // 128 MiB
  float* cov     = (float*)bump((size_t)B_ * 513 * COVLD * sizeof(float)); // ~65 MiB
  float* xty     = (float*)bump((size_t)B_ * XTYLD * sizeof(float));
  float* yc      = (float*)bump((size_t)B_ * NC_ * sizeof(float));
  float* meanstd = (float*)bump((size_t)B_ * 2 * sizeof(float));
  _Float16* wt   = (_Float16*)bump((size_t)3 * H_ * H_ * sizeof(_Float16));
  _Float16* crT  = act0;  // act0 is free after the head GEMM; reuse for cr^T

  k_stats<<<B_, 256, 0, stream>>>(yct, yc, meanstd);
  k_wconv<<<(3 * H_ * H_) / 256, 256, 0, stream>>>(W1, W2, Wr, wt);
  k_layer0<<<(int)(((long long)MROWS * H_) / 256), 256, 0, stream>>>(xc, xt, W0, b0, act0);

  dim3 gg(H_ / 64, MROWS / 128);  // (8, 1024)
  k_gemm_f16<<<gg, 256, 0, stream>>>(act0, wt,               b1, act1, 1);
  k_gemm_f16<<<gg, 256, 0, stream>>>(act1, wt + 1 * H_ * H_, b2, act0, 1);
  k_gemm_f16<<<gg, 256, 0, stream>>>(act0, wt + 2 * H_ * H_, br, act1, 0);

  k_transpose<<<(int)(((long long)B_ * H_ * NC_) / 256), 256, 0, stream>>>(act1, crT);
  k_syrk<<<dim3(8, 4, B_), 256, 0, stream>>>(crT, cov);
  k_border<<<dim3(3, B_), 256, 0, stream>>>(act1, yc, cov, xty, lnv);
  k_chol_solve<<<B_, 256, 0, stream>>>(cov, xty);
  k_pred<<<dim3(4, B_), 256, 0, stream>>>(act1, xty, meanstd, out);
}
